// TreeStructuredEncoder_44753559224522
// MI455X (gfx1250) — compile-verified, matrix operand order checked by reading, not run on hardware
//
#include <hip/hip_runtime.h>
#include <hip/hip_bf16.h>

typedef __attribute__((ext_vector_type(16))) __bf16        v16bf;
typedef __attribute__((ext_vector_type(8)))  float         v8f;
typedef __attribute__((ext_vector_type(4)))  unsigned int  v4u;

union Frag { v4u u[2]; v16bf v; };

#define MROWS    64      // rows per workgroup
#define LDA      776     // 768 + 8 bf16 pad -> row pitch 1552B (16B multiple)
#define DM       256
#define KTOT     768
#define OPSTRIDE 16383   // op_ids row stride (n_leaf - 1)

// ---------------------------------------------------------------------------
// prep: tok_emb -> bf16 ; W[0:768,:] -> bf16 N-major transpose
// ---------------------------------------------------------------------------
__global__ __launch_bounds__(256) void prep_convert(
    const float* __restrict__ tok_emb, const float* __restrict__ W,
    __hip_bfloat16* __restrict__ tokB, __hip_bfloat16* __restrict__ Wt)
{
    int i = blockIdx.x * 256 + threadIdx.x;
    if (i < 1024 * DM) tokB[i] = __float2bfloat16(tok_emb[i]);
    if (i < DM * KTOT) {
        int nIdx = i / KTOT, k = i % KTOT;
        Wt[i] = __float2bfloat16(W[k * DM + nIdx]);   // Wt[n][k] = W[k][n]
    }
}

// per-level bias: b + dvec@(W3+W5) + il@W4 + ir@W6
__global__ __launch_bounds__(256) void prep_bias(
    const float* __restrict__ dep_emb, const float* __restrict__ idx_emb,
    const float* __restrict__ W, const float* __restrict__ bvec,
    float* __restrict__ bias)
{
    int d = blockIdx.x;          // level 0..13, dvec = dep_emb[d+1]
    int j = threadIdx.x;         // output column
    const float* dv = dep_emb + (size_t)(d + 1) * DM;
    float s = bvec[j];
    for (int k = 0; k < DM; ++k) {
        s += dv[k] * (W[(size_t)(3 * DM + k) * DM + j] + W[(size_t)(5 * DM + k) * DM + j]);
        s += idx_emb[k]      * W[(size_t)(4 * DM + k) * DM + j];
        s += idx_emb[DM + k] * W[(size_t)(6 * DM + k) * DM + j];
    }
    bias[(size_t)d * DM + j] = s;
}

// leaf init: emb0 = bf16(tok_emb[leaf_ids] + dep_emb[14])
__global__ __launch_bounds__(256) void leaf_kernel(
    const int* __restrict__ leaf_ids, const float* __restrict__ tok_emb,
    const float* __restrict__ dep_emb, __hip_bfloat16* __restrict__ emb0)
{
    long i = (long)blockIdx.x * 256 + threadIdx.x;   // over 8*16384*256
    int  j = (int)(i & 255);
    long r = i >> 8;
    int id = leaf_ids[r];
    emb0[i] = __float2bfloat16(tok_emb[(size_t)id * DM + j] + dep_emb[14 * DM + j]);
}

// ---------------------------------------------------------------------------
// level kernel: h = relu([e_op|left|right] @ W[0:768] + bias) ; emb = LN(h)
// 256 threads = 8 waves; WG computes 64 rows x 256 cols; wave owns 2 N-tiles.
// ---------------------------------------------------------------------------
__global__ __launch_bounds__(256) void level_kernel(
    const __hip_bfloat16* __restrict__ embIn,   // (B*2n, 256) bf16
    __hip_bfloat16*       __restrict__ embOut,  // (B*n, 256) bf16
    float*                __restrict__ outF,    // d_out (used when isLast)
    const int*            __restrict__ op_ids,  // (8, 16383)
    const __hip_bfloat16* __restrict__ tokB,    // (1024, 256) bf16
    const __hip_bfloat16* __restrict__ Wt,      // (256, 768) bf16 N-major
    const float*          __restrict__ biasLvl, // 256
    const float*          __restrict__ gamma,
    const float*          __restrict__ beta,
    int n, int shift, int totalRows, int isLast)
{
    extern __shared__ char smem[];
    unsigned short* As  = (unsigned short*)smem;            // MROWS x LDA bf16
    float* lsum = (float*)(smem + MROWS * LDA * 2);
    float* lss  = lsum + MROWS;

    const int tid  = threadIdx.x;
    const int lane = tid & 31;
    const int wv   = tid >> 5;
    const int rowBase = blockIdx.x * MROWS;

    if (tid < MROWS) { lsum[tid] = 0.0f; lss[tid] = 0.0f; }

    // ---- cooperative A-tile load: 4 threads per row, 16B chunks ----
    {
        const int row  = tid >> 2;     // 0..63
        const int tsub = tid & 3;
        int gr = rowBase + row;
        if (gr >= totalRows) gr = totalRows - 1;           // clamp (EXEC must stay full for WMMA)
        const int b    = gr >> shift;
        const int node = gr & (n - 1);
        const int opid = op_ids[(size_t)b * OPSTRIDE + (n - 1) + node];
        const v4u* s0 = (const v4u*)(tokB  + (size_t)opid * DM);           // e_op
        const v4u* s1 = (const v4u*)(embIn + (size_t)(2 * gr) * DM);       // left
        const v4u* s2 = (const v4u*)(embIn + (size_t)(2 * gr + 1) * DM);   // right
        v4u* d0 = (v4u*)(As + row * LDA);
        v4u* d1 = (v4u*)(As + row * LDA + 256);
        v4u* d2 = (v4u*)(As + row * LDA + 512);
        #pragma unroll
        for (int c = 0; c < 8; ++c) {
            int off = c * 4 + tsub;                        // 0..31 (512B / seg)
            d0[off] = s0[off]; d1[off] = s1[off]; d2[off] = s2[off];
        }
    }
    __syncthreads();

    // ---- WMMA main loop: K=768 in 24 steps of 32 ----
    const int m     = lane & 15;       // A row within 16
    const int kh    = lane >> 4;       // lane-half -> K sub-pattern
    const int colLo = lane & 15;
    const int col0  = wv * 32 + colLo; // wave's two N-tiles
    const int col1  = col0 + 16;

    v8f acc[4][2];
    {
        v8f z = {};
        #pragma unroll
        for (int mt = 0; mt < 4; ++mt) { acc[mt][0] = z; acc[mt][1] = z; }
    }

    const unsigned short* w0 = (const unsigned short*)(Wt + (size_t)col0 * KTOT);
    const unsigned short* w1 = (const unsigned short*)(Wt + (size_t)col1 * KTOT);

    for (int kt = 0; kt < 24; ++kt) {
        const int k0 = kt * 32 + kh * 8;   // lanes 0-15: K 0..7 / 16..23 ; 16-31: 8..15 / 24..31
        const int k1 = k0 + 16;
        Frag bf0, bf1;
        bf0.u[0] = *(const v4u*)(w0 + k0);
        bf0.u[1] = *(const v4u*)(w0 + k1);
        bf1.u[0] = *(const v4u*)(w1 + k0);
        bf1.u[1] = *(const v4u*)(w1 + k1);
        if (kt + 1 < 24) {                                 // global_prefetch_b8 next K slice
            __builtin_prefetch(w0 + k0 + 32, 0, 0);
            __builtin_prefetch(w1 + k0 + 32, 0, 0);
        }
        #pragma unroll
        for (int mt = 0; mt < 4; ++mt) {
            const unsigned short* ar = As + (mt * 16 + m) * LDA;
            Frag af;
            af.u[0] = *(const v4u*)(ar + k0);
            af.u[1] = *(const v4u*)(ar + k1);
            acc[mt][0] = __builtin_amdgcn_wmma_f32_16x16x32_bf16(
                false, af.v, false, bf0.v, (short)0, acc[mt][0], false, false);
            acc[mt][1] = __builtin_amdgcn_wmma_f32_16x16x32_bf16(
                false, af.v, false, bf1.v, (short)0, acc[mt][1], false, false);
        }
    }

    // ---- bias + relu, layernorm partial sums (16-lane butterfly + LDS atomics) ----
    const float bias0 = biasLvl[col0];
    const float bias1 = biasLvl[col1];
    #pragma unroll
    for (int mt = 0; mt < 4; ++mt) {
        #pragma unroll
        for (int j = 0; j < 8; ++j) {      // C layout: VGPR j = row j (lanes 0-15) / j+8 (16-31)
            float v0 = fmaxf(acc[mt][0][j] + bias0, 0.0f);
            float v1 = fmaxf(acc[mt][1][j] + bias1, 0.0f);
            acc[mt][0][j] = v0;
            acc[mt][1][j] = v1;
            float s  = v0 + v1;
            float ss = v0 * v0 + v1 * v1;
            #pragma unroll
            for (int o = 1; o < 16; o <<= 1) {             // stays within 16-lane half
                s  += __shfl_xor(s,  o, 32);
                ss += __shfl_xor(ss, o, 32);
            }
            if (colLo == 0) {
                int row = mt * 16 + (kh ? 8 : 0) + j;
                atomicAdd(&lsum[row], s);                  // ds_add_f32
                atomicAdd(&lss[row],  ss);
            }
        }
    }
    __syncthreads();

    // ---- normalize + scale/shift + store ----
    const float g0 = gamma[col0], g1 = gamma[col1];
    const float be0 = beta[col0], be1 = beta[col1];
    #pragma unroll
    for (int mt = 0; mt < 4; ++mt) {
        #pragma unroll
        for (int j = 0; j < 8; ++j) {
            const int row = mt * 16 + (kh ? 8 : 0) + j;
            const int gr  = rowBase + row;
            if (gr >= totalRows) continue;
            const float mu  = lsum[row] * (1.0f / 256.0f);
            const float var = lss[row] * (1.0f / 256.0f) - mu * mu;
            const float rs  = rsqrtf(var + 1e-5f);
            const float o0 = (acc[mt][0][j] - mu) * rs * g0 + be0;
            const float o1 = (acc[mt][1][j] - mu) * rs * g1 + be1;
            const size_t base = (size_t)gr * DM;           // b*n+node == gr
            if (isLast) {
                outF[base + col0] = o0;
                outF[base + col1] = o1;
            } else {
                embOut[base + col0] = __float2bfloat16(o0);
                embOut[base + col1] = __float2bfloat16(o1);
            }
        }
    }
}

// ---------------------------------------------------------------------------
extern "C" void kernel_launch(void* const* d_in, const int* in_sizes, int n_in,
                              void* d_out, int out_size, void* d_ws, size_t ws_size,
                              hipStream_t stream) {
    const int*   leaf_ids = (const int*)  d_in[0];   // (8, 16384)
    const int*   op_ids   = (const int*)  d_in[1];   // (8, 16383)
    const float* tok_emb  = (const float*)d_in[2];   // (1024, 256)
    const float* dep_emb  = (const float*)d_in[3];   // (15, 256)
    const float* idx_emb  = (const float*)d_in[4];   // (2, 256)
    const float* W        = (const float*)d_in[5];   // (1792, 256)
    const float* bvec     = (const float*)d_in[6];   // (256,)
    const float* gamma    = (const float*)d_in[7];
    const float* beta     = (const float*)d_in[8];
    float* out = (float*)d_out;                      // (8, 256)

    // workspace carve-up
    char* ws = (char*)d_ws;
    __hip_bfloat16* tokB = (__hip_bfloat16*)ws;  ws += (size_t)1024 * DM * 2;   // 512 KB
    __hip_bfloat16* Wt   = (__hip_bfloat16*)ws;  ws += (size_t)DM * KTOT * 2;   // 384 KB
    float*          bias = (float*)ws;           ws += (size_t)14 * DM * 4;     // 14 KB
    __hip_bfloat16* embA = (__hip_bfloat16*)ws;  ws += (size_t)8 * 16384 * DM * 2; // 64 MB
    __hip_bfloat16* embB = (__hip_bfloat16*)ws;                                    // 32 MB

    prep_convert<<<1024, 256, 0, stream>>>(tok_emb, W, tokB, Wt);
    prep_bias<<<14, 256, 0, stream>>>(dep_emb, idx_emb, W, bvec, bias);
    leaf_kernel<<<131072, 256, 0, stream>>>(leaf_ids, tok_emb, dep_emb, embA);

    const size_t smemBytes = (size_t)MROWS * LDA * 2 + (size_t)MROWS * 4 * 2;

    __hip_bfloat16* curIn  = embA;
    __hip_bfloat16* curOut = embB;
    for (int d = 13; d >= 0; --d) {
        const int n = 1 << d;
        const int totalRows = 8 * n;
        const int grid = (totalRows + MROWS - 1) / MROWS;
        level_kernel<<<grid, 256, smemBytes, stream>>>(
            curIn, curOut, out, op_ids, tokB, Wt, bias + (size_t)d * DM,
            gamma, beta, n, d, totalRows, (d == 0) ? 1 : 0);
        __hip_bfloat16* t = curIn; curIn = curOut; curOut = t;
    }
}
